// FineTracker_73254962201195
// MI455X (gfx1250) — matrix operand order repaired and checked by reading, not compiled, not run on hardware
//
#include <hip/hip_runtime.h>
#include <hip/hip_bf16.h>
#include <math.h>

typedef __attribute__((ext_vector_type(16))) _Float16 v16h;
typedef __attribute__((ext_vector_type(8)))  _Float16 h8;
typedef __attribute__((ext_vector_type(8)))  float    v8f;
typedef __attribute__((ext_vector_type(4)))  float    f4;

// ---------------- problem constants (match setup_inputs) ----------------
constexpr int SS    = 16;    // frames S (= T)
constexpr int CC    = 128;   // feature dim
constexpr int HH    = 64;
constexpr int WW    = 64;
constexpr int NN    = 1024;  // points
constexpr int ITERS = 4;
constexpr int TDIM  = 312;
constexpr int KPAD  = 320;   // TDIM padded to multiple of 32 for WMMA K-loop
constexpr int HID   = 384;
constexpr int LAYERS= 4;
constexpr int NHEAD = 8;
constexpr int DHEAD = 48;
constexpr int LRR   = 147;   // 3 levels * 49 offsets
constexpr int MTOK  = NN * SS; // 16384 token rows

__device__ __forceinline__ float wave_sum32(float v) {
  #pragma unroll
  for (int o = 16; o; o >>= 1) v += __shfl_xor(v, o, 32);
  return v;
}
__device__ __forceinline__ int clampi(int v, int lo, int hi) {
  return v < lo ? lo : (v > hi ? hi : v);
}
__device__ __forceinline__ float gelu_tanh(float x) {
  float x3 = x * x * x;
  return 0.5f * x * (1.0f + tanhf(0.7978845608028654f * (x + 0.044715f * x3)));
}

// ---------------- pyramid 2x average-pool ----------------
__global__ void ft_pyr_down(const float* __restrict__ src, float* __restrict__ dst,
                            int hs, int ws) {
  int idx = blockIdx.x * blockDim.x + threadIdx.x;
  int hd = hs >> 1, wd = ws >> 1;
  int total = SS * CC * hd * wd;
  if (idx >= total) return;
  int x = idx % wd;
  int y = (idx / wd) % hd;
  int sc = idx / (wd * hd);
  const float* p = src + ((size_t)sc * hs + 2 * y) * ws + 2 * x;
  dst[idx] = 0.25f * (p[0] + p[1] + p[ws] + p[ws + 1]);
}

// ---------------- init: q_feat, track_feats broadcast, coords ----------------
__global__ void ft_init(const float* __restrict__ xys, const float* __restrict__ fmaps,
                        float* __restrict__ track, float* __restrict__ coords,
                        float* __restrict__ qfeat_out) {
  int n = blockIdx.x;
  int c = threadIdx.x;             // 128 threads
  float fx = xys[n * 2 + 0], fy = xys[n * 2 + 1];
  float x0f = floorf(fx), y0f = floorf(fy);
  float wx = fx - x0f, wy = fy - y0f;
  int x0 = clampi((int)x0f, 0, WW - 1), x1 = clampi((int)x0f + 1, 0, WW - 1);
  int y0 = clampi((int)y0f, 0, HH - 1), y1 = clampi((int)y0f + 1, 0, HH - 1);
  const float* p = fmaps + (size_t)c * HH * WW;   // s=0
  float v = p[y0 * WW + x0] * (1 - wx) * (1 - wy) + p[y0 * WW + x1] * wx * (1 - wy)
          + p[y1 * WW + x0] * (1 - wx) * wy       + p[y1 * WW + x1] * wx * wy;
  qfeat_out[(size_t)n * CC + c] = v;
  #pragma unroll
  for (int s = 0; s < SS; ++s) track[((size_t)s * NN + n) * CC + c] = v;
  if (c < 32) {
    int s = c >> 1, j = c & 1;
    coords[((size_t)s * NN + n) * 2 + j] = xys[n * 2 + j];
  }
}

// ---------------- sampled_pos_emb: bilinear of analytic sincos_2d ----------------
__global__ void ft_spe(const float* __restrict__ xys, float* __restrict__ spe) {
  int n = blockIdx.x;
  int c = threadIdx.x;             // 320 threads, guard
  if (c >= TDIM) return;
  float fx = xys[n * 2 + 0], fy = xys[n * 2 + 1];
  float x0f = floorf(fx), y0f = floorf(fy);
  float wx = fx - x0f, wy = fy - y0f;
  int x0 = clampi((int)x0f, 0, WW - 1), x1 = clampi((int)x0f + 1, 0, WW - 1);
  int y0 = clampi((int)y0f, 0, HH - 1), y1 = clampi((int)y0f + 1, 0, HH - 1);
  int band = c / 78;               // 0:sin(x) 1:cos(x) 2:sin(y) 3:cos(y)
  int i = c - band * 78;
  float om = powf(10000.0f, -(float)i / 78.0f);
  float e00, e01, e10, e11;
  if (band < 2) {
    float g0 = (band & 1) ? cosf((float)x0 * om) : sinf((float)x0 * om);
    float g1 = (band & 1) ? cosf((float)x1 * om) : sinf((float)x1 * om);
    e00 = g0; e01 = g1; e10 = g0; e11 = g1;        // depends on x only
  } else {
    float g0 = (band & 1) ? cosf((float)y0 * om) : sinf((float)y0 * om);
    float g1 = (band & 1) ? cosf((float)y1 * om) : sinf((float)y1 * om);
    e00 = g0; e01 = g0; e10 = g1; e11 = g1;        // depends on y only
  }
  float v = e00 * (1 - wx) * (1 - wy) + e01 * wx * (1 - wy)
          + e10 * (1 - wx) * wy       + e11 * wx * wy;
  spe[(size_t)n * TDIM + c] = v;
}

// ---------------- fused correlation sampling ----------------
// fcorr[(n*S+s)*147 + j] = track[s,n,:] . bilinear(fm_lvl[s], c/2^l + delta_j) / sqrt(C)
__global__ __launch_bounds__(256) void ft_fcorr(const float* __restrict__ coords,
    const float* __restrict__ track, const float* __restrict__ fm0,
    const float* __restrict__ fm1, const float* __restrict__ fm2,
    float* __restrict__ fcorr) {
  int sn = blockIdx.x;                  // s*N + n
  int s = sn / NN, n = sn % NN;
  int wave = threadIdx.x >> 5, lane = threadIdx.x & 31;
  float cx = coords[(size_t)sn * 2 + 0];
  float cy = coords[(size_t)sn * 2 + 1];
  f4 tf = *(const f4*)(track + (size_t)sn * CC + (lane << 2));
  const float rs = 0.08838834764831845f;   // 1/sqrt(128)
  for (int j = wave; j < LRR; j += 8) {
    int lvl = j / 49, k = j % 49;
    int hl = HH >> lvl;
    const float* fm = (lvl == 0) ? fm0 : ((lvl == 1) ? fm1 : fm2);
    float sc = 1.0f / (float)(1 << lvl);
    float fx = cx * sc + (float)(k / 7 - 3);   // delta[:,0] applied to x (ref quirk)
    float fy = cy * sc + (float)(k % 7 - 3);
    float x0f = floorf(fx), y0f = floorf(fy);
    float wx = fx - x0f, wy = fy - y0f;
    int x0 = clampi((int)x0f, 0, hl - 1), x1 = clampi((int)x0f + 1, 0, hl - 1);
    int y0 = clampi((int)y0f, 0, hl - 1), y1 = clampi((int)y0f + 1, 0, hl - 1);
    float w00 = (1 - wx) * (1 - wy), w01 = wx * (1 - wy);
    float w10 = (1 - wx) * wy,       w11 = wx * wy;
    const float* base = fm + ((size_t)s * CC + (lane << 2)) * hl * hl;
    float acc = 0.f;
    #pragma unroll
    for (int cc = 0; cc < 4; ++cc) {
      const float* p = base + (size_t)cc * hl * hl;
      float v = w00 * p[y0 * hl + x0] + w01 * p[y0 * hl + x1]
              + w10 * p[y1 * hl + x0] + w11 * p[y1 * hl + x1];
      acc += ((float)tf[cc]) * v;
    }
    acc = wave_sum32(acc);
    if (lane == 0) fcorr[((size_t)n * SS + s) * LRR + j] = acc * rs;
  }
}

// ---------------- token assembly: flows_emb | fcorr | track | pad, + pos emb, -> f16 ----------------
__global__ void ft_token(const float* __restrict__ coords, const float* __restrict__ track,
                         const float* __restrict__ fcorr, const float* __restrict__ spe,
                         _Float16* __restrict__ x16) {
  int row = blockIdx.x;               // n*S + s
  int n = row / SS, s = row % SS;
  int c = threadIdx.x;                // 320
  if (c >= KPAD) return;
  size_t o = (size_t)row * KPAD + c;
  if (c >= TDIM) { x16[o] = (_Float16)0.f; return; }
  float base;
  if (c < 34) {
    float fx = coords[((size_t)s * NN + n) * 2 + 0] - coords[(size_t)n * 2 + 0];
    float fy = coords[((size_t)s * NN + n) * 2 + 1] - coords[(size_t)n * 2 + 1];
    if (c < 32) {
      float v = (c < 16) ? fx : fy;
      int j = (c & 15) >> 1;
      float a = v * 125.0f * (float)j;   // div_j = (2j)*(1000/16)
      base = (c & 1) ? cosf(a) : sinf(a);
    } else {
      base = ((c == 32) ? fx : fy) * 0.5f;   // / STABLE_SCALE
    }
  } else if (c < 34 + LRR) {
    base = fcorr[(size_t)row * LRR + (c - 34)];
  } else if (c < 34 + LRR + CC) {
    base = track[((size_t)s * NN + n) * CC + (c - 181)];
  } else {
    base = 0.f;                        // pad
  }
  x16[o] = (_Float16)(base + spe[(size_t)n * TDIM + c]);
}

// ---------------- layernorm (no affine) f32 -> f16 ----------------
__global__ __launch_bounds__(256) void ft_ln(const float* __restrict__ x,
                                             _Float16* __restrict__ out) {
  int wave = threadIdx.x >> 5, lane = threadIdx.x & 31;
  int row = blockIdx.x * 8 + wave;
  const float* p = x + (size_t)row * HID;
  float v[12];
  float sum = 0.f;
  #pragma unroll
  for (int i = 0; i < 12; ++i) { v[i] = p[lane + i * 32]; sum += v[i]; }
  sum = wave_sum32(sum);
  float mean = sum * (1.0f / 384.0f);
  float q = 0.f;
  #pragma unroll
  for (int i = 0; i < 12; ++i) { float d = v[i] - mean; q += d * d; }
  q = wave_sum32(q);
  float rstd = rsqrtf(q * (1.0f / 384.0f) + 1e-6f);
  _Float16* o = out + (size_t)row * HID;
  #pragma unroll
  for (int i = 0; i < 12; ++i) o[lane + i * 32] = (_Float16)((v[i] - mean) * rstd);
}

// ---------------- WMMA GEMM: C[M x Nn] = A[M x K](f16) * Bt[Nn x K](f16, pre-transposed) ----------------
// Block tile 256x64, 8 waves, each wave 32x64 (2 A-frags share 4 B-frags -> 8 WMMA/k-tile).
// Double-buffered LDS; one barrier per k-tile; all staging via b128.
// flags: 1=bias, 2=gelu, 4=residual-add from Cf, 8=write f16, 16=write f32
template <int F>
__global__ __launch_bounds__(256) void ft_gemm(const _Float16* __restrict__ A,
    const _Float16* __restrict__ Bt, const float* __restrict__ bias,
    float* __restrict__ Cf, _Float16* __restrict__ Ch, int K, int Nn) {
  constexpr int AST = 48, BST = 48;     // LDS strides (halfs); 96B rows keep 16B alignment
  __shared__ _Float16 As[2][256 * AST];
  __shared__ _Float16 Bs[2][64 * BST];
  int tid = threadIdx.x;
  int wave = tid >> 5, lane = tid & 31;
  int bm = blockIdx.y * 256;
  int bn = blockIdx.x * 64;
  v8f acc[2][4];
  #pragma unroll
  for (int mi = 0; mi < 2; ++mi)
    #pragma unroll
    for (int j = 0; j < 4; ++j)
      #pragma unroll
      for (int r = 0; r < 8; ++r) acc[mi][j][r] = 0.f;

  // staging assignments: A row per thread (32 halfs = 4xb128), B 8 halfs per thread
  int arow = tid;                       // 0..255
  int bcol = tid >> 2;                  // 0..63
  int bkof = (tid & 3) * 8;             // 0,8,16,24
  const _Float16* aptr = A + (size_t)(bm + arow) * K;
  const _Float16* bptr = Bt + (size_t)(bn + bcol) * K + bkof;

  // prologue: stage k-tile 0
  h8 ar0 = *(const h8*)(aptr + 0);
  h8 ar1 = *(const h8*)(aptr + 8);
  h8 ar2 = *(const h8*)(aptr + 16);
  h8 ar3 = *(const h8*)(aptr + 24);
  h8 br  = *(const h8*)bptr;
  *(h8*)&As[0][arow * AST + 0]  = ar0;
  *(h8*)&As[0][arow * AST + 8]  = ar1;
  *(h8*)&As[0][arow * AST + 16] = ar2;
  *(h8*)&As[0][arow * AST + 24] = ar3;
  *(h8*)&Bs[0][bcol * BST + bkof] = br;
  __syncthreads();

  int nk = K >> 5;
  int mrow0 = wave * 32 + (lane & 15);  // first 16-row strip for this wave
  int akb   = (lane >> 4) * 8;
  int bkb   = (lane >> 4) * 8;
  for (int t = 0; t < nk; ++t) {
    int cur = t & 1, nxt = cur ^ 1;
    bool more = (t + 1 < nk);
    if (more) {                         // global loads for next tile fly under WMMAs
      const _Float16* ap = aptr + (t + 1) * 32;
      ar0 = *(const h8*)(ap + 0);
      ar1 = *(const h8*)(ap + 8);
      ar2 = *(const h8*)(ap + 16);
      ar3 = *(const h8*)(ap + 24);
      br  = *(const h8*)(bptr + (t + 1) * 32);
    }
    // A fragments: 16x32, lanes 0-15 M=lane K[0..7|16..23], lanes 16-31 K[8..15|24..31]
    union { v16h v; h8 h[2]; } af0, af1;
    af0.h[0] = *(const h8*)&As[cur][mrow0 * AST + akb];
    af0.h[1] = *(const h8*)&As[cur][mrow0 * AST + akb + 16];
    af1.h[0] = *(const h8*)&As[cur][(mrow0 + 16) * AST + akb];
    af1.h[1] = *(const h8*)&As[cur][(mrow0 + 16) * AST + akb + 16];
    #pragma unroll
    for (int j = 0; j < 4; ++j) {
      union { v16h v; h8 h[2]; } bf;
      int nc = j * 16 + (lane & 15);
      bf.h[0] = *(const h8*)&Bs[cur][nc * BST + bkb];
      bf.h[1] = *(const h8*)&Bs[cur][nc * BST + bkb + 16];
      acc[0][j] = __builtin_amdgcn_wmma_f32_16x16x32_f16(
          false, af0.v, false, bf.v, (short)0, acc[0][j], false, false);
      acc[1][j] = __builtin_amdgcn_wmma_f32_16x16x32_f16(
          false, af1.v, false, bf.v, (short)0, acc[1][j], false, false);
    }
    if (more) {
      *(h8*)&As[nxt][arow * AST + 0]  = ar0;
      *(h8*)&As[nxt][arow * AST + 8]  = ar1;
      *(h8*)&As[nxt][arow * AST + 16] = ar2;
      *(h8*)&As[nxt][arow * AST + 24] = ar3;
      *(h8*)&Bs[nxt][bcol * BST + bkof] = br;
    }
    __syncthreads();
  }

  // epilogue: C layout VGPR r -> lanes0-15 M=r, lanes16-31 M=r+8
  #pragma unroll
  for (int mi = 0; mi < 2; ++mi) {
    #pragma unroll
    for (int j = 0; j < 4; ++j) {
      #pragma unroll
      for (int r = 0; r < 8; ++r) {
        int m = bm + wave * 32 + mi * 16 + r + (lane >> 4) * 8;
        int nn = bn + j * 16 + (lane & 15);
        float v = acc[mi][j][r];
        if (F & 1) v += bias[nn];
        if (F & 2) v = gelu_tanh(v);
        size_t off = (size_t)m * Nn + nn;
        if (F & 4) v += Cf[off];
        if (F & 16) Cf[off] = v;
        if (F & 8) Ch[off] = (_Float16)v;
      }
    }
  }
}

// ---------------- attention: per sequence (T=16), heads=8, dh=48 ----------------
__global__ __launch_bounds__(128) void ft_attn(const _Float16* __restrict__ q,
    const _Float16* __restrict__ kv, _Float16* __restrict__ o) {
  __shared__ _Float16 skv[16 * 768];
  int seq = blockIdx.x;
  int tid = threadIdx.x;
  const _Float16* src = kv + (size_t)seq * 16 * 768;
  for (int i = tid * 8; i < 16 * 768; i += 128 * 8)
    *(h8*)&skv[i] = *(const h8*)&src[i];
  __syncthreads();
  int head = tid >> 4, t = tid & 15;
  const _Float16* qr = q + ((size_t)seq * 16 + t) * HID + head * DHEAD;
  float qv[DHEAD];
  #pragma unroll
  for (int d = 0; d < DHEAD; ++d) qv[d] = (float)qr[d];
  float sc[16];
  float mx = -1e30f;
  #pragma unroll
  for (int s2 = 0; s2 < 16; ++s2) {
    const _Float16* kr = &skv[s2 * 768 + head * DHEAD];
    float a = 0.f;
    #pragma unroll
    for (int d = 0; d < DHEAD; ++d) a += qv[d] * (float)kr[d];
    a *= 0.14433756729740645f;   // 48^-0.5
    sc[s2] = a;
    mx = fmaxf(mx, a);
  }
  float sum = 0.f;
  #pragma unroll
  for (int s2 = 0; s2 < 16; ++s2) { sc[s2] = expf(sc[s2] - mx); sum += sc[s2]; }
  float inv = 1.0f / sum;
  _Float16* orow = o + ((size_t)seq * 16 + t) * HID + head * DHEAD;
  #pragma unroll
  for (int d = 0; d < DHEAD; ++d) {
    float a = 0.f;
    #pragma unroll
    for (int s2 = 0; s2 < 16; ++s2)
      a += sc[s2] * (float)skv[s2 * 768 + 384 + head * DHEAD + d];
    orow[d] = (_Float16)(a * inv);
  }
}

// ---------------- head projection + coordinate update + pred output (wave/row) ----------------
__global__ __launch_bounds__(256) void ft_head(const float* __restrict__ tok,
    const float* __restrict__ hw, const float* __restrict__ hb,
    const float* __restrict__ xys, float* __restrict__ coords,
    float* __restrict__ preds) {
  int wave = threadIdx.x >> 5, lane = threadIdx.x & 31;
  int idx = blockIdx.x * 8 + wave;      // token row = n*S + s
  int n = idx / SS, s = idx % SS;
  const float* r = tok + (size_t)idx * HID;
  float d0 = 0.f, d1 = 0.f;
  #pragma unroll
  for (int i = 0; i < 12; ++i) {
    int k = lane + i * 32;
    float v = r[k];
    d0 += v * hw[k * 2 + 0];
    d1 += v * hw[k * 2 + 1];
  }
  d0 = wave_sum32(d0);
  d1 = wave_sum32(d1);
  if (lane == 0) {
    size_t co = ((size_t)s * NN + n) * 2;
    float nx, ny;
    if (s == 0) { nx = xys[n * 2 + 0]; ny = xys[n * 2 + 1]; }
    else {
      nx = coords[co + 0] + (d0 + hb[0]) * 0.5f;   // / STABLE_SCALE
      ny = coords[co + 1] + (d1 + hb[1]) * 0.5f;
    }
    coords[co + 0] = nx; coords[co + 1] = ny;
    preds[co + 0] = nx;  preds[co + 1] = ny;
  }
}

// ---------------- resample track features at new coords ----------------
__global__ void ft_resample(const float* __restrict__ coords,
                            const float* __restrict__ fmaps, float* __restrict__ track) {
  int sn = blockIdx.x;                  // s*N + n
  int c = threadIdx.x;                  // 128
  int s = sn / NN;
  float fx = coords[(size_t)sn * 2 + 0];
  float fy = coords[(size_t)sn * 2 + 1];
  float x0f = floorf(fx), y0f = floorf(fy);
  float wx = fx - x0f, wy = fy - y0f;
  int x0 = clampi((int)x0f, 0, WW - 1), x1 = clampi((int)x0f + 1, 0, WW - 1);
  int y0 = clampi((int)y0f, 0, HH - 1), y1 = clampi((int)y0f + 1, 0, HH - 1);
  const float* p = fmaps + ((size_t)s * CC + c) * HH * WW;
  float v = p[y0 * WW + x0] * (1 - wx) * (1 - wy) + p[y0 * WW + x1] * wx * (1 - wy)
          + p[y1 * WW + x0] * (1 - wx) * wy       + p[y1 * WW + x1] * wx * wy;
  track[(size_t)sn * CC + c] = v;
}

// ---------------- weight conversion: f32 row-major [K x N] -> f16 transposed [N x K] ----------------
__global__ void ft_cvt_t(const float* __restrict__ src, _Float16* __restrict__ dst,
                         int K, int Nc) {
  int i = blockIdx.x * blockDim.x + threadIdx.x;   // over K*Nc (src order, coalesced read)
  if (i >= K * Nc) return;
  int k = i / Nc, n = i % Nc;
  dst[(size_t)n * K + k] = (_Float16)src[i];
}
__global__ void ft_cvt_t_inw(const float* __restrict__ src, _Float16* __restrict__ dst) {
  int i = blockIdx.x * blockDim.x + threadIdx.x;   // over HID*KPAD (dst order)
  if (i >= HID * KPAD) return;
  int n = i / KPAD, k = i % KPAD;
  dst[i] = (k < TDIM) ? (_Float16)src[(size_t)k * HID + n] : (_Float16)0.f;
}

// =====================================================================
extern "C" void kernel_launch(void* const* d_in, const int* in_sizes, int n_in,
                              void* d_out, int out_size, void* d_ws, size_t ws_size,
                              hipStream_t stream) {
  const float* xys   = (const float*)d_in[0];
  const float* fmaps = (const float*)d_in[1];
  const float* in_w  = (const float*)d_in[2];
  const float* in_b  = (const float*)d_in[3];
  const float* awq   = (const float*)d_in[4];
  const float* abq   = (const float*)d_in[5];
  const float* awkv  = (const float*)d_in[6];
  const float* abkv  = (const float*)d_in[7];
  const float* awo   = (const float*)d_in[8];
  const float* abo   = (const float*)d_in[9];
  const float* mw1   = (const float*)d_in[10];
  const float* mb1   = (const float*)d_in[11];
  const float* mw2   = (const float*)d_in[12];
  const float* mb2   = (const float*)d_in[13];
  const float* hw    = (const float*)d_in[14];
  const float* hb    = (const float*)d_in[15];
  // d_in[16] = iters (device scalar) -> fixed 4 per setup

  char* ws = (char*)d_ws;
  size_t off = 0;
  auto alloc = [&](size_t bytes) -> char* {
    char* p = ws + off;
    off = (off + bytes + 255) & ~(size_t)255;
    return p;
  };
  float*     pyr1   = (float*)alloc(sizeof(float) * SS * CC * 32 * 32);
  float*     pyr2   = (float*)alloc(sizeof(float) * SS * CC * 16 * 16);
  float*     track  = (float*)alloc(sizeof(float) * SS * NN * CC);
  float*     coords = (float*)alloc(sizeof(float) * SS * NN * 2);
  float*     spe    = (float*)alloc(sizeof(float) * NN * TDIM);
  float*     fcorrb = (float*)alloc(sizeof(float) * (size_t)NN * SS * LRR);
  _Float16*  x16    = (_Float16*)alloc(sizeof(_Float16) * (size_t)MTOK * KPAD);
  float*     tokf   = (float*)alloc(sizeof(float) * (size_t)MTOK * HID);
  _Float16*  h16    = (_Float16*)alloc(sizeof(_Float16) * (size_t)MTOK * HID);
  _Float16*  q16    = (_Float16*)alloc(sizeof(_Float16) * (size_t)MTOK * HID);
  _Float16*  kv16   = (_Float16*)alloc(sizeof(_Float16) * (size_t)MTOK * 2 * HID);
  _Float16*  ao16   = (_Float16*)alloc(sizeof(_Float16) * (size_t)MTOK * HID);
  _Float16*  hid16  = (_Float16*)alloc(sizeof(_Float16) * (size_t)MTOK * 4 * HID);
  _Float16*  inw16  = (_Float16*)alloc(sizeof(_Float16) * HID * KPAD);        // [N x K]
  _Float16*  wq16   = (_Float16*)alloc(sizeof(_Float16) * LAYERS * HID * HID);
  _Float16*  wkv16  = (_Float16*)alloc(sizeof(_Float16) * LAYERS * HID * 2 * HID);
  _Float16*  wo16   = (_Float16*)alloc(sizeof(_Float16) * LAYERS * HID * HID);
  _Float16*  w116   = (_Float16*)alloc(sizeof(_Float16) * LAYERS * HID * 4 * HID);
  _Float16*  w216   = (_Float16*)alloc(sizeof(_Float16) * LAYERS * 4 * HID * HID);

  float* out       = (float*)d_out;
  float* preds     = out;                                 // (ITERS,1,S,N,2)
  float* track_out = out + (size_t)ITERS * SS * NN * 2;   // (1,S,N,C)
  float* qfeat_out = track_out + (size_t)SS * NN * CC;    // (1,N,C)

  // ---- setup ----
  ft_pyr_down<<<(SS * CC * 32 * 32 + 255) / 256, 256, 0, stream>>>(fmaps, pyr1, 64, 64);
  ft_pyr_down<<<(SS * CC * 16 * 16 + 255) / 256, 256, 0, stream>>>(pyr1, pyr2, 32, 32);
  ft_init<<<NN, 128, 0, stream>>>(xys, fmaps, track, coords, qfeat_out);
  ft_spe<<<NN, 320, 0, stream>>>(xys, spe);

  // weights: f32 [K x N] -> f16 transposed [N x K] (per layer)
  ft_cvt_t_inw<<<(HID * KPAD + 255) / 256, 256, 0, stream>>>(in_w, inw16);
  for (int l = 0; l < LAYERS; ++l) {
    ft_cvt_t<<<(HID * HID + 255) / 256, 256, 0, stream>>>(
        awq + (size_t)l * HID * HID, wq16 + (size_t)l * HID * HID, HID, HID);
    ft_cvt_t<<<(HID * 2 * HID + 255) / 256, 256, 0, stream>>>(
        awkv + (size_t)l * HID * 2 * HID, wkv16 + (size_t)l * HID * 2 * HID, HID, 2 * HID);
    ft_cvt_t<<<(HID * HID + 255) / 256, 256, 0, stream>>>(
        awo + (size_t)l * HID * HID, wo16 + (size_t)l * HID * HID, HID, HID);
    ft_cvt_t<<<(HID * 4 * HID + 255) / 256, 256, 0, stream>>>(
        mw1 + (size_t)l * HID * 4 * HID, w116 + (size_t)l * HID * 4 * HID, HID, 4 * HID);
    ft_cvt_t<<<(4 * HID * HID + 255) / 256, 256, 0, stream>>>(
        mw2 + (size_t)l * 4 * HID * HID, w216 + (size_t)l * 4 * HID * HID, 4 * HID, HID);
  }

  // ---- iterations ----
  for (int it = 0; it < ITERS; ++it) {
    ft_fcorr<<<SS * NN, 256, 0, stream>>>(coords, track, fmaps, pyr1, pyr2, fcorrb);
    ft_token<<<MTOK, 320, 0, stream>>>(coords, track, fcorrb, spe, x16);

    // tok = x @ in_w + in_b
    ft_gemm<17><<<dim3(HID / 64, MTOK / 256), 256, 0, stream>>>(
        x16, inw16, in_b, tokf, nullptr, KPAD, HID);

    for (int l = 0; l < LAYERS; ++l) {
      ft_ln<<<MTOK / 8, 256, 0, stream>>>(tokf, h16);
      ft_gemm<9><<<dim3(HID / 64, MTOK / 256), 256, 0, stream>>>(
          h16, wq16 + (size_t)l * HID * HID, abq + l * HID, nullptr, q16, HID, HID);
      ft_gemm<9><<<dim3(2 * HID / 64, MTOK / 256), 256, 0, stream>>>(
          h16, wkv16 + (size_t)l * HID * 2 * HID, abkv + l * 2 * HID, nullptr, kv16,
          HID, 2 * HID);
      ft_attn<<<NN, 128, 0, stream>>>(q16, kv16, ao16);
      ft_gemm<21><<<dim3(HID / 64, MTOK / 256), 256, 0, stream>>>(
          ao16, wo16 + (size_t)l * HID * HID, abo + l * HID, tokf, nullptr, HID, HID);
      ft_ln<<<MTOK / 8, 256, 0, stream>>>(tokf, h16);
      ft_gemm<11><<<dim3(4 * HID / 64, MTOK / 256), 256, 0, stream>>>(
          h16, w116 + (size_t)l * HID * 4 * HID, mb1 + l * 4 * HID, nullptr, hid16,
          HID, 4 * HID);
      ft_gemm<21><<<dim3(HID / 64, MTOK / 256), 256, 0, stream>>>(
          hid16, w216 + (size_t)l * 4 * HID * HID, mb2 + l * HID, tokf, nullptr,
          4 * HID, HID);
    }

    ft_head<<<MTOK / 8, 256, 0, stream>>>(
        tokf, hw, hb, xys, coords, preds + (size_t)it * SS * NN * 2);
    ft_resample<<<SS * NN, 128, 0, stream>>>(coords, fmaps, track);
  }

  hipMemcpyAsync(track_out, track, sizeof(float) * SS * NN * CC,
                 hipMemcpyDeviceToDevice, stream);
}
